// OptimizedGNNEncoder_31267361915476
// MI455X (gfx1250) — matrix operand order repaired and checked by reading, not compiled, not run on hardware
//
#include <hip/hip_runtime.h>
#include <hip/hip_bf16.h>
#include <math.h>

typedef __attribute__((ext_vector_type(16))) _Float16 v16h;
typedef __attribute__((ext_vector_type(8)))  float    v8f;
typedef __attribute__((ext_vector_type(4)))  float    v4f;
typedef __attribute__((ext_vector_type(4)))  _Float16 v4h;

#define N_NODES 50000
#define N_EDGES 800000
#define F_IN    8
#define H_DIM   64
#define K_CL    16
#define L_OUT   32
#define B_GR    50
#define TAU     1.0f
#define EPS_LN  1e-5f

// ---------------- workspace layout (floats) ----------------
#define WS_MH     ((size_t)0)
#define WS_MV     (WS_MH + (size_t)N_NODES * 16)
#define WS_H1     (WS_MV + (size_t)N_NODES * 2)
#define WS_H2     (WS_H1 + (size_t)N_NODES * H_DIM)
#define WS_DENOM  (WS_H2 + (size_t)N_NODES * H_DIM)
#define WS_NUM    (WS_DENOM + (size_t)B_GR * K_CL)
#define WS_MU     (WS_NUM + (size_t)B_GR * K_CL * H_DIM)
#define WS_ENT    (WS_MU + (size_t)B_GR * K_CL * 2)
#define WS_POOLED (WS_ENT + 1)

// ---------------- output layout (floats) ----------------
#define OUT_LATENT ((size_t)0)
#define OUT_S      ((size_t)B_GR * K_CL * L_OUT)
#define OUT_ENT    (OUT_S + (size_t)N_NODES * K_CL)
#define OUT_MU     (OUT_ENT + 1)

__device__ __forceinline__ v4h cvt4(v4f f) { return __builtin_convertvector(f, v4h); }

// ---------------- WMMA fragment helpers (gfx1250 wave32 layouts) ----------------
// A (16x32 f16): lanes 0-15 -> M=lane, K in {0..7,16..23}; lanes 16-31 -> K in {8..15,24..31}
__device__ __forceinline__ v16h frag_a(const _Float16* A, int lda, int kc, int lane) {
    const int m  = lane & 15;
    const int kb = kc * 32 + ((lane >> 4) << 3);
    const _Float16* p = A + m * lda + kb;
    v16h r;
#pragma unroll
    for (int i = 0; i < 8; ++i) { r[i] = p[i]; r[8 + i] = p[16 + i]; }
    return r;
}
// B (32x16 f16) from LDS weights stored transposed (N-major, K contiguous):
// lane holds column N = lane&15; lanes 0-15 K in {0..7,16..23}, lanes 16-31 K in {8..15,24..31}
__device__ __forceinline__ v16h frag_b(const _Float16* Wt, int ldk, int kc, int tile, int lane) {
    const int n  = tile * 16 + (lane & 15);
    const int kb = kc * 32 + ((lane >> 4) << 3);
    const _Float16* p = Wt + n * ldk + kb;
    v16h r;
#pragma unroll
    for (int i = 0; i < 8; ++i) { r[i] = p[i]; r[8 + i] = p[16 + i]; }
    return r;
}
#define WMMA(a, b, c) __builtin_amdgcn_wmma_f32_16x16x32_f16(false, (a), false, (b), (short)0, (c), false, false)

// ================= edge MLP: phi_e + phi_v, scatter m_h/m_v =================
template <int CIN, int DPAD>
__global__ __launch_bounds__(128) void edge_mlp_kernel(
    const float* __restrict__ feat, int fstride,
    const float* __restrict__ x,
    const int* __restrict__ esrc, const int* __restrict__ edst,
    const float* __restrict__ w1,  const float* __restrict__ b1,   // D x 32
    const float* __restrict__ w2,  const float* __restrict__ b2,   // 32 x 16
    const float* __restrict__ wv1, const float* __restrict__ bv1,  // D x 16
    const float* __restrict__ wv2, const float* __restrict__ bv2,  // 16 x 1
    float* __restrict__ m_h, float* __restrict__ m_v)
{
    constexpr int D  = 2 * CIN + 2;
    constexpr int KC = DPAD / 32;

    __shared__ __align__(16) _Float16 sW1t[32][DPAD];
    __shared__ __align__(16) _Float16 sW2t[16][32];
    __shared__ __align__(16) _Float16 sWv1t[16][DPAD];
    __shared__ float sWv2[16];
    __shared__ float sB1[32], sB2[16], sBv1[16], sBv2s;
    __shared__ __align__(16) _Float16 sA[4][16][DPAD];
    __shared__ __align__(16) _Float16 sH1[4][16][32];
    __shared__ __align__(16) _Float16 sHv[4][16][16];
    __shared__ int   sDst[4][16];
    __shared__ float sRel[4][16][2];

    const int tid = threadIdx.x;
    for (int i = tid; i < 32 * DPAD; i += 128) {
        int n = i / DPAD, k = i % DPAD;
        sW1t[n][k] = (k < D) ? (_Float16)w1[k * 32 + n] : (_Float16)0.f;
    }
    for (int i = tid; i < 16 * 32; i += 128) {
        int n = i / 32, k = i % 32;
        sW2t[n][k] = (_Float16)w2[k * 16 + n];
    }
    for (int i = tid; i < 16 * DPAD; i += 128) {
        int n = i / DPAD, k = i % DPAD;
        sWv1t[n][k] = (k < D) ? (_Float16)wv1[k * 16 + n] : (_Float16)0.f;
    }
    if (tid < 32) sB1[tid] = b1[tid];
    if (tid < 16) { sB2[tid] = b2[tid]; sBv1[tid] = bv1[tid]; sWv2[tid] = wv2[tid]; }
    if (tid == 0) sBv2s = bv2[0];
    __syncthreads();

    const int wave = tid >> 5, lane = tid & 31;
    const int half = lane >> 4, m = lane & 15;
    const int e = (blockIdx.x * 4 + wave) * 16 + m;
    const bool valid = (e < N_EDGES);
    const int si = valid ? esrc[e] : 0;
    const int di = valid ? edst[e] : 0;

    {   // gather tile: row m = [x[dst](CIN), x[src](CIN), dist_sq, dot_vr, 0-pad]
        const float* frow = feat + (size_t)(half ? si : di) * fstride;
        const int off = half ? CIN : 0;
        const v4f z4 = {0.f, 0.f, 0.f, 0.f};
#pragma unroll
        for (int c = 0; c < CIN; c += 4) {
            v4f f = valid ? *(const v4f*)(frow + c) : z4;
            *(v4h*)&sA[wave][m][off + c] = cvt4(f);
        }
        if (half == 0) {
            // pos/vel are x[:,0:2] and x[:,2:4] -> one 16B row load
            v4f pvs = valid ? *(const v4f*)(x + (size_t)si * F_IN) : z4;
            v4f pvd = valid ? *(const v4f*)(x + (size_t)di * F_IN) : z4;
            float px = pvs[0] - pvd[0], py = pvs[1] - pvd[1];
            float vx = pvs[2] - pvd[2], vy = pvs[3] - pvd[3];
            float dist = px * px + py * py;
            float dv   = vx * px + vy * py;
            sA[wave][m][2 * CIN]     = (_Float16)dist;
            sA[wave][m][2 * CIN + 1] = (_Float16)dv;
#pragma unroll
            for (int c = D; c < DPAD; ++c) sA[wave][m][c] = (_Float16)0.f;
            sDst[wave][m]    = valid ? di : -1;
            sRel[wave][m][0] = px; sRel[wave][m][1] = py;
        }
    }
    __syncthreads();

    // phi_e.l1 [16xDPAD]x[DPADx32] and phi_v.l1 [16xDPAD]x[DPADx16]
    v8f acc0 = {}, acc1 = {}, accv = {};
#pragma unroll
    for (int kc = 0; kc < KC; ++kc) {
        v16h a = frag_a(&sA[wave][0][0], DPAD, kc, lane);
        acc0 = WMMA(a, frag_b(&sW1t[0][0],  DPAD, kc, 0, lane), acc0);
        acc1 = WMMA(a, frag_b(&sW1t[0][0],  DPAD, kc, 1, lane), acc1);
        accv = WMMA(a, frag_b(&sWv1t[0][0], DPAD, kc, 0, lane), accv);
    }
    const int mrow = (lane >> 4) * 8;
    const int n    = lane & 15;
#pragma unroll
    for (int v = 0; v < 8; ++v) {   // bias + SiLU, write back as f16 tiles
        float h0 = acc0[v] + sB1[n];
        float h1 = acc1[v] + sB1[16 + n];
        float hv = accv[v] + sBv1[n];
        sH1[wave][mrow + v][n]      = (_Float16)(h0 / (1.f + __expf(-h0)));
        sH1[wave][mrow + v][16 + n] = (_Float16)(h1 / (1.f + __expf(-h1)));
        sHv[wave][mrow + v][n]      = (_Float16)(hv / (1.f + __expf(-hv)));
    }
    __syncthreads();

    // phi_e.l2 [16x32]x[32x16] -> m_h_e, scatter-add to m_h[dst]
    v8f acc2 = {};
    {
        v16h a = frag_a(&sH1[wave][0][0], 32, 0, lane);
        acc2 = WMMA(a, frag_b(&sW2t[0][0], 32, 0, 0, lane), acc2);
    }
#pragma unroll
    for (int v = 0; v < 8; ++v) {
        int row  = mrow + v;
        int dsti = sDst[wave][row];
        if (dsti >= 0) atomicAdd(&m_h[(size_t)dsti * 16 + n], acc2[v] + sB2[n]);
    }

    // phi_v.l2 (16->1) per edge, scatter m_v_e = val * rel_pos
    if (lane < 16) {
        float a = sBv2s;
#pragma unroll
        for (int k = 0; k < 16; ++k) a += (float)sHv[wave][lane][k] * sWv2[k];
        int dsti = sDst[wave][lane];
        if (dsti >= 0) {
            atomicAdd(&m_v[(size_t)dsti * 2 + 0], a * sRel[wave][lane][0]);
            atomicAdd(&m_v[(size_t)dsti * 2 + 1], a * sRel[wave][lane][1]);
        }
    }
}

// ================= node update: phi_h + shortcut + ReLU + LayerNorm =================
template <int CIN, int KPAD, bool HAS_SC>
__global__ __launch_bounds__(128) void node_mlp_kernel(
    const float* __restrict__ feat,
    const float* __restrict__ m_h, const float* __restrict__ m_v,
    const float* __restrict__ w1,  const float* __restrict__ b1,   // Dh x 16
    const float* __restrict__ w2,  const float* __restrict__ b2,   // 16 x 64
    const float* __restrict__ scw, const float* __restrict__ scb,  // CIN x 64 (opt)
    const float* __restrict__ lng, const float* __restrict__ lnb,
    float* __restrict__ hout)
{
    constexpr int Dh = CIN + 17;
    constexpr int KC = KPAD / 32;

    __shared__ __align__(16) _Float16 sW1t[16][KPAD];
    __shared__ __align__(16) _Float16 sW2t[64][32];
    __shared__ float sB1[16], sB2[64];
    __shared__ float sSC[CIN][64];
    __shared__ float sSCb[64];
    __shared__ float sG[64], sBeta[64];
    __shared__ __align__(16) _Float16 sA[4][16][KPAD];
    __shared__ __align__(16) _Float16 sH[4][16][32];
    __shared__ __align__(16) float sOut[4][16][64];

    const int tid = threadIdx.x;
    for (int i = tid; i < 16 * KPAD; i += 128) {
        int nn = i / KPAD, k = i % KPAD;
        sW1t[nn][k] = (k < Dh) ? (_Float16)w1[k * 16 + nn] : (_Float16)0.f;
    }
    for (int i = tid; i < 64 * 32; i += 128) {
        int nn = i / 32, k = i % 32;
        sW2t[nn][k] = (k < 16) ? (_Float16)w2[k * 64 + nn] : (_Float16)0.f;
    }
    if (HAS_SC) {
        for (int i = tid; i < CIN * 64; i += 128) sSC[i / 64][i % 64] = scw[i];
        if (tid < 64) sSCb[tid] = scb[tid];
    }
    if (tid < 16) sB1[tid] = b1[tid];
    if (tid < 64) { sB2[tid] = b2[tid]; sG[tid] = lng[tid]; sBeta[tid] = lnb[tid]; }
    for (int i = tid; i < 4 * 16 * 32; i += 128) (&sH[0][0][0])[i] = (_Float16)0.f;
    __syncthreads();

    const int wave = tid >> 5, lane = tid & 31;
    const int m = lane & 15;
    const int node = (blockIdx.x * 4 + wave) * 16 + m;
    const bool valid = node < N_NODES;

    if (lane < 16) {   // gather [x(CIN), m_h(16), |m_v|(1), 0-pad]
        const v4f z4 = {0.f, 0.f, 0.f, 0.f};
        const float* f  = feat + (size_t)node * CIN;
        const float* mh = m_h + (size_t)node * 16;
#pragma unroll
        for (int c = 0; c < CIN; c += 4)
            *(v4h*)&sA[wave][m][c] = cvt4(valid ? *(const v4f*)(f + c) : z4);
#pragma unroll
        for (int c = 0; c < 16; c += 4)
            *(v4h*)&sA[wave][m][CIN + c] = cvt4(valid ? *(const v4f*)(mh + c) : z4);
        float v0 = valid ? m_v[(size_t)node * 2 + 0] : 0.f;
        float v1 = valid ? m_v[(size_t)node * 2 + 1] : 0.f;
        sA[wave][m][CIN + 16] = (_Float16)sqrtf(v0 * v0 + v1 * v1 + 1e-12f);
#pragma unroll
        for (int c = Dh; c < KPAD; ++c) sA[wave][m][c] = (_Float16)0.f;
    }
    __syncthreads();

    v8f accm = {};
#pragma unroll
    for (int kc = 0; kc < KC; ++kc) {
        v16h a = frag_a(&sA[wave][0][0], KPAD, kc, lane);
        accm = WMMA(a, frag_b(&sW1t[0][0], KPAD, kc, 0, lane), accm);
    }
    const int mrow = (lane >> 4) * 8;
    const int n    = lane & 15;
#pragma unroll
    for (int v = 0; v < 8; ++v) {
        float h = accm[v] + sB1[n];
        sH[wave][mrow + v][n] = (_Float16)(h / (1.f + __expf(-h)));
    }
    __syncthreads();

    v8f o[4] = {};
    {
        v16h a = frag_a(&sH[wave][0][0], 32, 0, lane);
#pragma unroll
        for (int t = 0; t < 4; ++t)
            o[t] = WMMA(a, frag_b(&sW2t[0][0], 32, 0, t, lane), o[t]);
    }
#pragma unroll
    for (int t = 0; t < 4; ++t) {
#pragma unroll
        for (int v = 0; v < 8; ++v) {
            int row = mrow + v;
            int col = t * 16 + n;
            int nd  = (blockIdx.x * 4 + wave) * 16 + row;
            float sc = 0.f;
            if (HAS_SC) {
                sc = sSCb[col];
                if (nd < N_NODES) {
                    const float* f = feat + (size_t)nd * CIN;
                    for (int c = 0; c < CIN; ++c) sc += f[c] * sSC[c][col];
                }
            } else {
                sc = (nd < N_NODES) ? feat[(size_t)nd * CIN + col] : 0.f;
            }
            sOut[wave][row][col] = fmaxf(sc + o[t][v] + sB2[col], 0.f);
        }
    }
    __syncthreads();

    if (lane < 16 && valid) {   // LayerNorm over 64
        float mu = 0.f;
        for (int c = 0; c < 64; ++c) mu += sOut[wave][m][c];
        mu *= (1.f / 64.f);
        float var = 0.f;
        for (int c = 0; c < 64; ++c) { float d = sOut[wave][m][c] - mu; var += d * d; }
        var *= (1.f / 64.f);
        float inv = rsqrtf(var + EPS_LN);
        float* orow = hout + (size_t)node * 64;
#pragma unroll
        for (int c = 0; c < 64; c += 4) {
            v4f r;
#pragma unroll
            for (int j = 0; j < 4; ++j)
                r[j] = (sOut[wave][m][c + j] - mu) * inv * sG[c + j] + sBeta[c + j];
            *(v4f*)(orow + c) = r;
        }
    }
}

// ================= pooling: softmax(h@Wp/tau) + segment atomics =================
__global__ __launch_bounds__(128) void pool_kernel(
    const float* __restrict__ h, const float* __restrict__ x,
    const int* __restrict__ batch,
    const float* __restrict__ pw, const float* __restrict__ pb,
    float* __restrict__ s_out,
    float* __restrict__ denom, float* __restrict__ num,
    float* __restrict__ mu, float* __restrict__ ent)
{
    __shared__ __align__(16) _Float16 sPWt[16][64];
    __shared__ float sPB[16];
    __shared__ __align__(16) _Float16 sHf[4][16][64];
    __shared__ __align__(16) float sS[4][16][16];
    __shared__ float sPos[4][16][2];
    __shared__ int sBt[4][16];

    const int tid = threadIdx.x;
    for (int i = tid; i < 16 * 64; i += 128)
        sPWt[i / 64][i % 64] = (_Float16)pw[(i % 64) * 16 + (i / 64)];
    if (tid < 16) sPB[tid] = pb[tid];
    __syncthreads();

    const int wave = tid >> 5, lane = tid & 31;
    const int m = lane & 15;
    const int node = (blockIdx.x * 4 + wave) * 16 + m;
    const bool valid = node < N_NODES;

    if (lane < 16) {
        const v4f z4 = {0.f, 0.f, 0.f, 0.f};
        const float* hr = h + (size_t)node * 64;
#pragma unroll
        for (int c = 0; c < 64; c += 4)
            *(v4h*)&sHf[wave][m][c] = cvt4(valid ? *(const v4f*)(hr + c) : z4);
        if (valid) {
            sPos[wave][m][0] = x[(size_t)node * F_IN + 0];
            sPos[wave][m][1] = x[(size_t)node * F_IN + 1];
            sBt[wave][m] = batch[node];
        } else {
            sPos[wave][m][0] = sPos[wave][m][1] = 0.f;
            sBt[wave][m] = -1;
        }
    }
    __syncthreads();

    v8f acc = {};
#pragma unroll
    for (int kc = 0; kc < 2; ++kc) {
        v16h a = frag_a(&sHf[wave][0][0], 64, kc, lane);
        acc = WMMA(a, frag_b(&sPWt[0][0], 64, kc, 0, lane), acc);
    }
    const int mrow = (lane >> 4) * 8;
    const int n    = lane & 15;
#pragma unroll
    for (int v = 0; v < 8; ++v)
        sS[wave][mrow + v][n] = (acc[v] + sPB[n]) * (1.f / TAU);
    __syncthreads();

    if (lane < 16 && valid) {   // per-node softmax over K=16 + entropy + s output
        float mx = -1e30f;
        for (int k = 0; k < 16; ++k) mx = fmaxf(mx, sS[wave][m][k]);
        float sv[16], sum = 0.f;
        for (int k = 0; k < 16; ++k) { sv[k] = __expf(sS[wave][m][k] - mx); sum += sv[k]; }
        float inv = 1.f / sum, entl = 0.f;
        float* srow = s_out + (size_t)node * 16;
#pragma unroll
        for (int k = 0; k < 16; k += 4) {
            v4f r;
#pragma unroll
            for (int j = 0; j < 4; ++j) {
                float s = sv[k + j] * inv;
                sS[wave][m][k + j] = s;
                r[j] = s;
                entl += s * __logf(s + 1e-9f);
            }
            *(v4f*)(srow + k) = r;
        }
        atomicAdd(ent, entl);
    }
    __syncthreads();

    for (int mm = 0; mm < 16; ++mm) {   // scatter num / denom / mu
        int bb = sBt[wave][mm];
        if (bb < 0) continue;
        for (int i = lane; i < 16 * 64; i += 32) {
            int k = i >> 6, c = i & 63;
            atomicAdd(&num[((size_t)bb * 16 + k) * 64 + c],
                      sS[wave][mm][k] * (float)sHf[wave][mm][c]);
        }
        if (lane < 16) atomicAdd(&denom[(size_t)bb * 16 + lane], sS[wave][mm][lane]);
        {
            int k = lane >> 1, ax = lane & 1;
            atomicAdd(&mu[((size_t)bb * 16 + k) * 2 + ax],
                      sS[wave][mm][k] * sPos[wave][mm][ax]);
        }
    }
}

// ================= finalize pooled / mu / entropy =================
__global__ void finalize_kernel(const float* __restrict__ num, const float* __restrict__ denom,
                                const float* __restrict__ mu_acc, const float* __restrict__ ent_acc,
                                float* __restrict__ pooled,
                                float* __restrict__ out_ent, float* __restrict__ out_mu)
{
    int i = blockIdx.x * blockDim.x + threadIdx.x;
    if (i < B_GR * K_CL * H_DIM) {
        int r = i / H_DIM;
        pooled[i] = num[i] / (denom[r] + 1e-8f);
    }
    if (i < B_GR * K_CL * 2) {
        int r = i >> 1;
        out_mu[i] = mu_acc[i] / (denom[r] + 1e-8f);
    }
    if (i == 0) out_ent[0] = -(ent_acc[0] / (float)N_NODES);
}

// ================= output MLP: (B*K) x 64 -> relu 32 -> 32 =================
__global__ __launch_bounds__(128) void outmlp_kernel(
    const float* __restrict__ pooled,
    const float* __restrict__ w1, const float* __restrict__ b1,   // 64 x 32
    const float* __restrict__ w2, const float* __restrict__ b2,   // 32 x 32
    float* __restrict__ latent)
{
    __shared__ __align__(16) _Float16 sW1t[32][64];
    __shared__ __align__(16) _Float16 sW2t[32][32];
    __shared__ float sB1[32], sB2[32];
    __shared__ __align__(16) _Float16 sA[4][16][64];
    __shared__ __align__(16) _Float16 sM[4][16][32];

    const int tid = threadIdx.x;
    for (int i = tid; i < 32 * 64; i += 128)
        sW1t[i / 64][i % 64] = (_Float16)w1[(i % 64) * 32 + (i / 64)];
    for (int i = tid; i < 32 * 32; i += 128)
        sW2t[i / 32][i % 32] = (_Float16)w2[(i % 32) * 32 + (i / 32)];
    if (tid < 32) { sB1[tid] = b1[tid]; sB2[tid] = b2[tid]; }
    __syncthreads();

    const int wave = tid >> 5, lane = tid & 31;
    const int m = lane & 15;
    const int row = (blockIdx.x * 4 + wave) * 16 + m;
    const bool valid = row < B_GR * K_CL;

    if (lane < 16) {
        const v4f z4 = {0.f, 0.f, 0.f, 0.f};
        const float* pr = pooled + (size_t)row * 64;
#pragma unroll
        for (int c = 0; c < 64; c += 4)
            *(v4h*)&sA[wave][m][c] = cvt4(valid ? *(const v4f*)(pr + c) : z4);
    }
    __syncthreads();

    v8f a0 = {}, a1 = {};
#pragma unroll
    for (int kc = 0; kc < 2; ++kc) {
        v16h a = frag_a(&sA[wave][0][0], 64, kc, lane);
        a0 = WMMA(a, frag_b(&sW1t[0][0], 64, kc, 0, lane), a0);
        a1 = WMMA(a, frag_b(&sW1t[0][0], 64, kc, 1, lane), a1);
    }
    const int mrow = (lane >> 4) * 8;
    const int n    = lane & 15;
#pragma unroll
    for (int v = 0; v < 8; ++v) {
        sM[wave][mrow + v][n]      = (_Float16)fmaxf(a0[v] + sB1[n], 0.f);
        sM[wave][mrow + v][16 + n] = (_Float16)fmaxf(a1[v] + sB1[16 + n], 0.f);
    }
    __syncthreads();

    v8f o0 = {}, o1 = {};
    {
        v16h a = frag_a(&sM[wave][0][0], 32, 0, lane);
        o0 = WMMA(a, frag_b(&sW2t[0][0], 32, 0, 0, lane), o0);
        o1 = WMMA(a, frag_b(&sW2t[0][0], 32, 0, 1, lane), o1);
    }
#pragma unroll
    for (int v = 0; v < 8; ++v) {
        int r = (blockIdx.x * 4 + wave) * 16 + mrow + v;
        if (r < B_GR * K_CL) {
            latent[(size_t)r * 32 + n]      = o0[v] + sB2[n];
            latent[(size_t)r * 32 + 16 + n] = o1[v] + sB2[16 + n];
        }
    }
}

__global__ void zero_kernel(float* __restrict__ p, size_t n) {
    size_t i = (size_t)blockIdx.x * blockDim.x + threadIdx.x;
    if (i < n) p[i] = 0.f;
}

// ================= host-side orchestration =================
#define PF(i) ((const float*)d_in[(i)])

extern "C" void kernel_launch(void* const* d_in, const int* in_sizes, int n_in,
                              void* d_out, int out_size, void* d_ws, size_t ws_size,
                              hipStream_t stream)
{
    (void)in_sizes; (void)n_in; (void)out_size; (void)ws_size;

    const float* x     = (const float*)d_in[0];
    const int*   eidx  = (const int*)d_in[1];
    const int*   batch = (const int*)d_in[2];
    const int*   esrc  = eidx;
    const int*   edst  = eidx + N_EDGES;
    // params (dict insertion order):
    // 3..6  gnn1.phi_e.{l1.w,l1.b,l2.w,l2.b}   7..10 gnn1.phi_v   11..14 gnn1.phi_h
    // 15,16 gnn1.sc.{w,b}
    // 17..20 gnn2.phi_e  21..24 gnn2.phi_v  25..28 gnn2.phi_h
    // 29,30 ln1.{g,b}  31,32 ln2.{g,b}  33,34 pool.{w,b}
    // 35..38 out.{l1.w,l1.b,l2.w,l2.b}

    float* ws     = (float*)d_ws;
    float* mh     = ws + WS_MH;
    float* mv     = ws + WS_MV;
    float* h1     = ws + WS_H1;
    float* h2     = ws + WS_H2;
    float* denom  = ws + WS_DENOM;
    float* num    = ws + WS_NUM;
    float* muacc  = ws + WS_MU;
    float* entacc = ws + WS_ENT;
    float* pooled = ws + WS_POOLED;

    float* out        = (float*)d_out;
    float* out_latent = out + OUT_LATENT;
    float* out_s      = out + OUT_S;
    float* out_ent    = out + OUT_ENT;
    float* out_mu     = out + OUT_MU;

#define ZERO(p, cnt) zero_kernel<<<(unsigned)(((cnt) + 255) / 256), 256, 0, stream>>>((p), (size_t)(cnt))
    ZERO(mh, (size_t)N_NODES * 16);
    ZERO(mv, (size_t)N_NODES * 2);
    ZERO(denom, B_GR * K_CL);
    ZERO(num, (size_t)B_GR * K_CL * H_DIM);
    ZERO(muacc, B_GR * K_CL * 2);
    ZERO(entacc, 1);

    const int eBlocks = (N_EDGES + 63) / 64;
    const int nBlocks = (N_NODES + 63) / 64;

    // layer 1: cin=8, D=18 -> pad 32
    edge_mlp_kernel<8, 32><<<eBlocks, 128, 0, stream>>>(
        x, F_IN, x, esrc, edst,
        PF(3), PF(4), PF(5), PF(6), PF(7), PF(8), PF(9), PF(10), mh, mv);
    node_mlp_kernel<8, 32, true><<<nBlocks, 128, 0, stream>>>(
        x, mh, mv, PF(11), PF(12), PF(13), PF(14), PF(15), PF(16), PF(29), PF(30), h1);

    ZERO(mh, (size_t)N_NODES * 16);
    ZERO(mv, (size_t)N_NODES * 2);

    // layer 2: cin=64, D=130 -> pad 160
    edge_mlp_kernel<64, 160><<<eBlocks, 128, 0, stream>>>(
        h1, H_DIM, x, esrc, edst,
        PF(17), PF(18), PF(19), PF(20), PF(21), PF(22), PF(23), PF(24), mh, mv);
    node_mlp_kernel<64, 96, false><<<nBlocks, 128, 0, stream>>>(
        h1, mh, mv, PF(25), PF(26), PF(27), PF(28), nullptr, nullptr, PF(31), PF(32), h2);

    // pooling + finalize + output MLP
    pool_kernel<<<nBlocks, 128, 0, stream>>>(
        h2, x, batch, PF(33), PF(34), out_s, denom, num, muacc, entacc);
    finalize_kernel<<<(B_GR * K_CL * H_DIM + 255) / 256, 256, 0, stream>>>(
        num, denom, muacc, entacc, pooled, out_ent, out_mu);
    outmlp_kernel<<<(B_GR * K_CL + 63) / 64, 128, 0, stream>>>(
        pooled, PF(35), PF(36), PF(37), PF(38), out_latent);
#undef ZERO
}
